// LSTM_1_52218212384942
// MI455X (gfx1250) — matrix operand order repaired and checked by reading, not compiled, not run on hardware
//
#include <hip/hip_runtime.h>
#include <hip/hip_bf16.h>

typedef __attribute__((ext_vector_type(16))) _Float16 v16h;
typedef __attribute__((ext_vector_type(8)))  float    v8f;

#define H   2048
#define SEQ 2048
#define G   128            // persistent workgroups; each owns H/G = 16 hidden units
#define TPB 256            // 8 waves: 4 gates x 2 K-halves
#define KT  (H / 32)       // 64 k-tiles of 32

// workspace layout (bytes)
#define SZ_W     ((size_t)4 * H * H * 2)          // one tiled fp16 weight matrix: 32 MiB
#define OFF_W0   ((size_t)0)
#define OFF_W1I  (SZ_W)
#define OFF_W1H  (2 * SZ_W)
#define OFF_H0H  (3 * SZ_W)                        // double-buffered fp16 h0: 2*H halfs
#define OFF_H1H  (OFF_H0H + (size_t)2 * H * 2)     // double-buffered fp16 h1
#define OFF_H1F  (OFF_H1H + (size_t)2 * H * 2)     // fp32 h1 for y
#define OFF_BAR  (OFF_H1F + (size_t)H * 4)         // barrier count + generation

// ---------------------------------------------------------------------------
// Pre-pass: convert a (4H x H) fp32 row-major matrix into fp16 tiles laid out
// in this file's contiguous k order for the WMMA A-operand (16x32 tile -> 32 lanes x 16 halfs):
//   lane<16 : row = lane,    K = {0..15}
//   lane>=16: row = lane-16, K = {16..31}
// so the hot loop is a single contiguous 32B load per lane per tile.
// ---------------------------------------------------------------------------
__global__ void __launch_bounds__(256) convert_w(const float* __restrict__ src,
                                                 _Float16* __restrict__ dst) {
  const int gwave = (blockIdx.x * 256 + threadIdx.x) >> 5;  // tile id (32768 per matrix)
  const int lane  = threadIdx.x & 31;
  const int tr = gwave / KT;          // tile-row (16 rows each), 0..511
  const int tk = gwave % KT;          // k-tile (32 cols each), 0..63
  const int row = tr * 16 + (lane & 15);
  const int colBase = tk * 32 + ((lane >> 4) * 16);
  v16h v;
#pragma unroll
  for (int e = 0; e < 16; ++e) {
    int col = colBase + e;
    v[e] = (_Float16)src[(size_t)row * H + col];
  }
  *(v16h*)(dst + (size_t)gwave * 512 + lane * 16) = v;
}

__global__ void init_state(_Float16* h0h, _Float16* h1h, int* bar) {
  int i = blockIdx.x * blockDim.x + threadIdx.x;
  if (i < 2 * H) { h0h[i] = (_Float16)0.f; h1h[i] = (_Float16)0.f; }
  if (i == 0) { bar[0] = 0; bar[1] = 0; }
}

__device__ __forceinline__ float sigf(float x) { return 1.f / (1.f + __expf(-x)); }

// device-wide generation barrier (persistent grid, no cooperative launch).
// Arrival is an RMW; the wait is a scoped plain load so 127 spinners don't
// serialize the L2 atomic unit against the gate-open store.
__device__ __forceinline__ void gbar(int* cnt, int* gen, int target) {
  __syncthreads();
  if (threadIdx.x == 0) {
    __threadfence();                       // release my h writes
    int prev = atomicAdd(cnt, 1);
    if (prev == G - 1) {
      atomicExch(cnt, 0);                  // reset before opening the gate
      __threadfence();
      __hip_atomic_store(gen, target, __ATOMIC_RELEASE, __HIP_MEMORY_SCOPE_AGENT);
    } else {
      while (__hip_atomic_load(gen, __ATOMIC_ACQUIRE, __HIP_MEMORY_SCOPE_AGENT) < target)
        __builtin_amdgcn_s_sleep(1);
      __threadfence();                     // acquire others' h writes
    }
  }
  __syncthreads();
}

// ---------------------------------------------------------------------------
// Persistent LSTM kernel: G workgroups, 2 grid barriers per timestep.
// Wave w (0..7): gate = w&3, khalf = w>>2 -> 16 gate rows over half of K.
// ---------------------------------------------------------------------------
__global__ void __launch_bounds__(TPB) lstm_persistent(
    const float* __restrict__ x, const int* __restrict__ mask,
    const float* __restrict__ wih0, const float* __restrict__ bih0,
    const float* __restrict__ bhh0, const float* __restrict__ bih1,
    const float* __restrict__ bhh1, const float* __restrict__ wffn,
    const float* __restrict__ bffn,
    const _Float16* __restrict__ W0, const _Float16* __restrict__ W1i,
    const _Float16* __restrict__ W1h,
    _Float16* __restrict__ h0h, _Float16* __restrict__ h1h,
    float* __restrict__ h1f, int* __restrict__ bar, float* __restrict__ out) {
  __shared__ float part[8][16];   // per-wave column-0 partial sums
  __shared__ float ypart[8];      // per-wave y partials
  __shared__ float ybc;           // broadcast y

  const int tid    = threadIdx.x;
  const int wg     = blockIdx.x;
  const int wave   = tid >> 5;
  const int lane   = tid & 31;
  const int laneHi = lane >> 4;
  const int gate   = wave & 3;
  const int khalf  = wave >> 2;

  const int tr = gate * (H / 16) + wg;                      // tile-row in 4H space
  const size_t aBase = ((size_t)tr * KT + (size_t)khalf * 32) * 512 + (size_t)lane * 16;
  const _Float16* a0  = W0  + aBase;
  const _Float16* a1i = W1i + aBase;
  const _Float16* a1h = W1h + aBase;
  // B operand: lanes 0-15 need h[kbase+0..15], lanes 16-31 need h[kbase+16..31]
  const int bOff = khalf * 1024 + laneHi * 16;

  // per-thread constants for the elementwise stage (hidden unit u)
  const int u = wg * 16 + (tid & 15);
  float bs0[4] = {0, 0, 0, 0}, wi0[4] = {0, 0, 0, 0}, bs1[4] = {0, 0, 0, 0};
  if (tid < 16) {
#pragma unroll
    for (int q = 0; q < 4; ++q) {
      int r = q * H + u;
      bs0[q] = bih0[r] + bhh0[r];
      wi0[q] = wih0[r];
      bs1[q] = bih1[r] + bhh1[r];
    }
  }
  float c0r = 0.f, c1r = 0.f, prev_y = 0.f;
  const float bf = bffn[0];
  int* cnt = bar;
  int* gen = bar + 1;
  int target = 1;

  for (int t = 0; t < SEQ; ++t) {
    const int p = t & 1;
    const _Float16* h0rd = h0h + p * H;
    _Float16*       h0wr = h0h + (1 - p) * H;
    const _Float16* h1rd = h1h + p * H;
    _Float16*       h1wr = h1h + (1 - p) * H;

    const float inp = (t == 0 || mask[t] != 0) ? x[t] : prev_y;

    // ---- layer 0: gates0 = w_hh0 @ h0_prev (column 0 of WMMA D) ----
    {
      v8f acc = {0.f, 0.f, 0.f, 0.f, 0.f, 0.f, 0.f, 0.f};
      const _Float16* ap = a0;
      const _Float16* bp = h0rd + bOff;
#pragma unroll 8
      for (int k = 0; k < 32; ++k) {
        v16h av = *(const v16h*)ap; ap += 512;
        v16h bv = *(const v16h*)bp; bp += 32;
        acc = __builtin_amdgcn_wmma_f32_16x16x32_f16(false, av, false, bv,
                                                     (short)0, acc, false, false);
      }
      if (lane == 0 || lane == 16) {
#pragma unroll
        for (int m = 0; m < 8; ++m) part[wave][laneHi * 8 + m] = acc[m];
      }
    }
    __syncthreads();
    if (tid < 16) {
      float gi = part[0][tid] + part[4][tid] + wi0[0] * inp + bs0[0];
      float gf = part[1][tid] + part[5][tid] + wi0[1] * inp + bs0[1];
      float gg = part[2][tid] + part[6][tid] + wi0[2] * inp + bs0[2];
      float go = part[3][tid] + part[7][tid] + wi0[3] * inp + bs0[3];
      float iv = sigf(gi), fv = sigf(gf), gv = tanhf(gg), ov = sigf(go);
      c0r = fv * c0r + iv * gv;
      float h0v = ov * tanhf(c0r);
      h0wr[u] = (_Float16)h0v;
    }
    gbar(cnt, gen, target++);   // h0(t) visible device-wide

    // ---- layer 1: gates1 = w_ih1 @ h0_new + w_hh1 @ h1_prev ----
    {
      v8f acc = {0.f, 0.f, 0.f, 0.f, 0.f, 0.f, 0.f, 0.f};
      const _Float16* ap = a1i;
      const _Float16* bp = h0wr + bOff;
#pragma unroll 8
      for (int k = 0; k < 32; ++k) {
        v16h av = *(const v16h*)ap; ap += 512;
        v16h bv = *(const v16h*)bp; bp += 32;
        acc = __builtin_amdgcn_wmma_f32_16x16x32_f16(false, av, false, bv,
                                                     (short)0, acc, false, false);
      }
      ap = a1h;
      bp = h1rd + bOff;
#pragma unroll 8
      for (int k = 0; k < 32; ++k) {
        v16h av = *(const v16h*)ap; ap += 512;
        v16h bv = *(const v16h*)bp; bp += 32;
        acc = __builtin_amdgcn_wmma_f32_16x16x32_f16(false, av, false, bv,
                                                     (short)0, acc, false, false);
      }
      if (lane == 0 || lane == 16) {
#pragma unroll
        for (int m = 0; m < 8; ++m) part[wave][laneHi * 8 + m] = acc[m];
      }
    }
    __syncthreads();
    if (tid < 16) {
      float gi = part[0][tid] + part[4][tid] + bs1[0];
      float gf = part[1][tid] + part[5][tid] + bs1[1];
      float gg = part[2][tid] + part[6][tid] + bs1[2];
      float go = part[3][tid] + part[7][tid] + bs1[3];
      float iv = sigf(gi), fv = sigf(gf), gv = tanhf(gg), ov = sigf(go);
      c1r = fv * c1r + iv * gv;
      float h1v = ov * tanhf(c1r);
      h1wr[u] = (_Float16)h1v;
      h1f[u]  = h1v;
    }
    gbar(cnt, gen, target++);   // h1(t) visible device-wide

    // ---- y = w_ffn @ h1 + b_ffn, computed redundantly per WG (no 3rd barrier)
    float s = 0.f;
#pragma unroll
    for (int j = tid; j < H; j += TPB) s += wffn[j] * h1f[j];
#pragma unroll
    for (int off = 16; off > 0; off >>= 1) s += __shfl_down(s, off, 32);
    if (lane == 0) ypart[wave] = s;
    __syncthreads();
    if (tid == 0) {
      float acc = ypart[0];
#pragma unroll
      for (int w = 1; w < 8; ++w) acc += ypart[w];
      ybc = acc + bf;
      if (wg == 0) out[t] = acc + bf;
    }
    __syncthreads();
    prev_y = ybc;
  }
}

extern "C" void kernel_launch(void* const* d_in, const int* in_sizes, int n_in,
                              void* d_out, int out_size, void* d_ws, size_t ws_size,
                              hipStream_t stream) {
  (void)in_sizes; (void)n_in; (void)out_size; (void)ws_size;
  const float* x    = (const float*)d_in[0];
  const int*   mask = (const int*)d_in[1];
  // d_in[2] = teacher_force_ratio (==1, deterministic teacher forcing) — unused
  const float* wih0 = (const float*)d_in[3];
  const float* whh0 = (const float*)d_in[4];
  const float* bih0 = (const float*)d_in[5];
  const float* bhh0 = (const float*)d_in[6];
  const float* wih1 = (const float*)d_in[7];
  const float* whh1 = (const float*)d_in[8];
  const float* bih1 = (const float*)d_in[9];
  const float* bhh1 = (const float*)d_in[10];
  const float* wffn = (const float*)d_in[11];
  const float* bffn = (const float*)d_in[12];

  char* ws = (char*)d_ws;
  _Float16* W0  = (_Float16*)(ws + OFF_W0);
  _Float16* W1i = (_Float16*)(ws + OFF_W1I);
  _Float16* W1h = (_Float16*)(ws + OFF_W1H);
  _Float16* h0h = (_Float16*)(ws + OFF_H0H);
  _Float16* h1h = (_Float16*)(ws + OFF_H1H);
  float*    h1f = (float*)(ws + OFF_H1F);
  int*      bar = (int*)(ws + OFF_BAR);
  float*    out = (float*)d_out;

  // one-time (per call) fp32 -> tiled fp16 weight swizzle; L2-resident afterwards
  convert_w<<<4096, 256, 0, stream>>>(whh0, W0);
  convert_w<<<4096, 256, 0, stream>>>(wih1, W1i);
  convert_w<<<4096, 256, 0, stream>>>(whh1, W1h);
  init_state<<<16, 256, 0, stream>>>(h0h, h1h, bar);

  lstm_persistent<<<G, TPB, 0, stream>>>(x, mask, wih0, bih0, bhh0, bih1, bhh1,
                                         wffn, bffn, W0, W1i, W1h,
                                         h0h, h1h, h1f, bar, out);
}
